// gru_d_18897856102812
// MI455X (gfx1250) — matrix operand order repaired and checked by reading, not compiled, not run on hardware
//
#include <hip/hip_runtime.h>
#include <hip/hip_bf16.h>
#include <math.h>

// ---------------- problem constants ----------------
constexpr int Bn = 2048, Tn = 128, Vn = 59, Hn = 256;
constexpr int K_IH = 128;   // 2V=118 padded to 128
constexpr int K_DH = 64;    // V=59 padded to 64
constexpr int G3H  = 3 * Hn; // 768

// ---------------- workspace layout (bytes) ----------------
constexpr size_t AMB_OFF = 0;                                   // bf16 [T][B][128]  (x_h | m | pad)
constexpr size_t AMB_SZ  = (size_t)Tn * Bn * K_IH * 2;          // 64 MB
constexpr size_t DB_OFF  = AMB_OFF + AMB_SZ;                    // bf16 [T][B][64]   (deltas | pad)
constexpr size_t DB_SZ   = (size_t)Tn * Bn * K_DH * 2;          // 32 MB
constexpr size_t WIH_OFF = DB_OFF + DB_SZ;                      // bf16 [768][128]
constexpr size_t WIH_SZ  = (size_t)G3H * K_IH * 2;
constexpr size_t WHH_OFF = WIH_OFF + WIH_SZ;                    // bf16 [768][256]
constexpr size_t WHH_SZ  = (size_t)G3H * Hn * 2;
constexpr size_t WDH_OFF = WHH_OFF + WHH_SZ;                    // bf16 [256][64]
constexpr size_t WDH_SZ  = (size_t)Hn * K_DH * 2;
constexpr size_t H_OFF   = WDH_OFF + WDH_SZ;                    // f32 [B][H]  recurrent state
constexpr size_t H_SZ    = (size_t)Bn * Hn * 4;
constexpr size_t HSF_OFF = H_OFF + H_SZ;                        // f32 [B][H]  h*gamma_h
constexpr size_t HSF_SZ  = (size_t)Bn * Hn * 4;
constexpr size_t HSB_OFF = HSF_OFF + HSF_SZ;                    // bf16 [B][H]
constexpr size_t HSB_SZ  = (size_t)Bn * Hn * 2;
constexpr size_t NUM_OFF = HSB_OFF + HSB_SZ;                    // f32 [T]
constexpr size_t DEN_OFF = NUM_OFF + 512;                       // f32 [T]
constexpr size_t WS_NEED = DEN_OFF + 512;

// ---------------- types / helpers ----------------
typedef __attribute__((ext_vector_type(16))) __bf16        v16bf;
typedef __attribute__((ext_vector_type(8)))  float         v8f;
typedef __attribute__((ext_vector_type(4)))  unsigned int  v4u;

__device__ __forceinline__ unsigned short f2bf(float f) {
    union { float f; unsigned int u; } c; c.f = f;
    unsigned int u = c.u;
    unsigned int r = (u + 0x7FFFu + ((u >> 16) & 1u)) >> 16;   // round-to-nearest-even
    return (unsigned short)r;
}

// A-matrix 16x32 bf16 fragment (ISA 7.12.2), row-major source [M][ldk]:
// lane holds row m; low lanes take K=[kb,kb+8) and [kb+16,kb+24),
// high lanes take K=[kb+8,kb+16) and [kb+24,kb+32)  -> two b128 loads.
__device__ __forceinline__ v16bf load_afrag(const unsigned short* base, int m, int ldk,
                                            int kbase, int lane) {
    union { v16bf v; v4u q[2]; } u;
    const unsigned short* row = base + (size_t)m * ldk + kbase + ((lane >= 16) ? 8 : 0);
    u.q[0] = *(const v4u*)(row);
    u.q[1] = *(const v4u*)(row + 16);
    return u.v;
}

// B-matrix 32x16 bf16 fragment, B[k][n] = W[n*ldk + k] (W row-major [N][K]):
// lane holds column n; low lanes K=[kb,kb+8) and [kb+16,kb+24),
// high lanes K=[kb+8,kb+16) and [kb+24,kb+32)  -> two b128 loads.
__device__ __forceinline__ v16bf load_bfrag(const unsigned short* w, int ncol, int ldk,
                                            int kbase, int lane) {
    union { v16bf v; v4u q[2]; } u;
    const unsigned short* row = w + (size_t)ncol * ldk + kbase + ((lane >= 16) ? 8 : 0);
    u.q[0] = *(const v4u*)(row);
    u.q[1] = *(const v4u*)(row + 16);
    return u.v;
}

__device__ __forceinline__ v8f wmma_bf16(v16bf a, v16bf b, v8f c) {
    return __builtin_amdgcn_wmma_f32_16x16x32_bf16(false, a, false, b, (short)0, c,
                                                   false, false);
}

__device__ __forceinline__ v8f zero8() {
    v8f z;
#pragma unroll
    for (int i = 0; i < 8; ++i) z[i] = 0.0f;
    return z;
}

__device__ __forceinline__ float sigmoidf_(float x) { return 1.0f / (1.0f + expf(-x)); }

// ---------------- kernel 0: convert weights to zero-padded bf16 ----------------
__global__ void weights_kernel(const float* __restrict__ Wih, const float* __restrict__ Whh,
                               const float* __restrict__ Wdh, unsigned short* __restrict__ wih_b,
                               unsigned short* __restrict__ whh_b, unsigned short* __restrict__ wdh_b) {
    int i = blockIdx.x * blockDim.x + threadIdx.x;
    if (i < G3H * K_IH) {
        int n = i >> 7, k = i & 127;
        wih_b[i] = (k < 2 * Vn) ? f2bf(Wih[n * (2 * Vn) + k]) : 0;
    }
    if (i < G3H * Hn) {
        whh_b[i] = f2bf(Whh[i]);
    }
    if (i < Hn * K_DH) {
        int n = i >> 6, k = i & 63;
        wdh_b[i] = (k < Vn) ? f2bf(Wdh[n * Vn + k]) : 0;
    }
}

// ---------------- kernel 1: elementwise forward-fill scan + imputation + loss parts ----------------
__global__ void scan_kernel(const float* __restrict__ x, const float* __restrict__ mask,
                            const float* __restrict__ deltas, const float* __restrict__ meanset,
                            const float* __restrict__ Wdx, const float* __restrict__ bdx,
                            float* __restrict__ out, unsigned short* __restrict__ amb,
                            unsigned short* __restrict__ db, float* __restrict__ num,
                            float* __restrict__ den) {
    int idx = blockIdx.x * blockDim.x + threadIdx.x;   // grid == B*V exactly
    int b = idx / Vn, v = idx % Vn;
    float wd = Wdx[v * Vn + v];
    float bx = bdx[v];
    float mu = meanset[v];
    float f = 0.0f;
    for (int t = 0; t < Tn; ++t) {
        size_t base = ((size_t)b * Tn + t) * Vn + v;     // [B,T,V]
        float xv = x[base], mv = mask[base], dv = deltas[base];
        f = mv * xv + (1.0f - mv) * f;
        float gx = expf(-fmaxf(0.0f, dv * wd + bx));
        float xu = gx * f + (1.0f - gx) * mu;
        float xh = mv * xv + (1.0f - mv) * xu;
        out[base] = xh;                                  // x_imp
        size_t arow = ((size_t)t * Bn + b) * K_IH;
        amb[arow + v]      = f2bf(xh);
        amb[arow + Vn + v] = f2bf(mv);
        db[((size_t)t * Bn + b) * K_DH + v] = f2bf(dv);
        float nm = fabsf(xv - xu) * mv;
        float dn = mv;
#pragma unroll
        for (int o = 16; o > 0; o >>= 1) {
            nm += __shfl_down(nm, o, 32);
            dn += __shfl_down(dn, o, 32);
        }
        if ((threadIdx.x & 31) == 0) {
            atomicAdd(&num[t], nm);
            atomicAdd(&den[t], dn);
        }
    }
}

// ---------------- kernel 2 (per step): gamma_h GEMM + h scaling ----------------
// tile grid: 128 (M) x 16 (N) = 2048 waves; one 16x16 tile per wave, K=64
__global__ void stepA_kernel(const unsigned short* __restrict__ db,
                             const unsigned short* __restrict__ wdh_b,
                             const float* __restrict__ bdh, const float* __restrict__ h,
                             float* __restrict__ hsf, unsigned short* __restrict__ hs_b, int t) {
    int wave = (blockIdx.x * blockDim.x + threadIdx.x) >> 5;
    int lane = threadIdx.x & 31;
    int tileM = wave >> 4, tileN = wave & 15;
    const unsigned short* A = db + (size_t)t * Bn * K_DH;
    int mA = tileM * 16 + (lane & 15);
    int nB = tileN * 16 + (lane & 15);

    v8f acc = zero8();
#pragma unroll
    for (int kb = 0; kb < K_DH; kb += 32) {
        v16bf a = load_afrag(A, mA, K_DH, kb, lane);
        v16bf b = load_bfrag(wdh_b, nB, K_DH, kb, lane);
        acc = wmma_bf16(a, b, acc);
    }
    int nOut = tileN * 16 + (lane & 15);
    int mBase = tileM * 16 + ((lane >= 16) ? 8 : 0);
    float bias = bdh[nOut];
#pragma unroll
    for (int r = 0; r < 8; ++r) {
        int mm = mBase + r;
        float g  = expf(-fmaxf(0.0f, acc[r] + bias));
        size_t o = (size_t)mm * Hn + nOut;
        float hv = h[o] * g;
        hsf[o]  = hv;
        hs_b[o] = f2bf(hv);
    }
}

// ---------------- kernel 3 (per step): gate GEMMs + GRU update ----------------
// tile grid: 128 (M) x 16 (N over H=256 gate-space) = 2048 waves
__global__ void stepB_kernel(const unsigned short* __restrict__ amb,
                             const unsigned short* __restrict__ hs_b,
                             const unsigned short* __restrict__ wih_b,
                             const unsigned short* __restrict__ whh_b,
                             const float* __restrict__ b_ih, const float* __restrict__ b_hh,
                             const float* __restrict__ hsf, float* __restrict__ h, int t) {
    int wave = (blockIdx.x * blockDim.x + threadIdx.x) >> 5;
    int lane = threadIdx.x & 31;
    int tileM = wave >> 4, tileN = wave & 15;
    int mA = tileM * 16 + (lane & 15);
    int nB = tileN * 16 + (lane & 15);

    v8f accR = zero8(), accZ = zero8(), accNx = zero8(), accNh = zero8();

    // gx part: [x_h,m] (K=128) x W_ih^T   (r/z fold into shared accumulators)
    const unsigned short* Ax = amb + (size_t)t * Bn * K_IH;
#pragma unroll
    for (int kb = 0; kb < K_IH; kb += 32) {
        v16bf a  = load_afrag(Ax, mA, K_IH, kb, lane);
        v16bf br = load_bfrag(wih_b, nB,            K_IH, kb, lane);
        v16bf bz = load_bfrag(wih_b, Hn + nB,       K_IH, kb, lane);
        v16bf bn = load_bfrag(wih_b, 2 * Hn + nB,   K_IH, kb, lane);
        accR  = wmma_bf16(a, br, accR);
        accZ  = wmma_bf16(a, bz, accZ);
        accNx = wmma_bf16(a, bn, accNx);
    }
    // gh part: (h*gamma_h) (K=256) x W_hh^T
#pragma unroll
    for (int kb = 0; kb < Hn; kb += 32) {
        v16bf a  = load_afrag(hs_b, mA, Hn, kb, lane);
        v16bf br = load_bfrag(whh_b, nB,          Hn, kb, lane);
        v16bf bz = load_bfrag(whh_b, Hn + nB,     Hn, kb, lane);
        v16bf bn = load_bfrag(whh_b, 2 * Hn + nB, Hn, kb, lane);
        accR  = wmma_bf16(a, br, accR);
        accZ  = wmma_bf16(a, bz, accZ);
        accNh = wmma_bf16(a, bn, accNh);
    }

    int c = tileN * 16 + (lane & 15);
    int mBase = tileM * 16 + ((lane >= 16) ? 8 : 0);
    float bihr = b_ih[c]          + b_hh[c];
    float bihz = b_ih[Hn + c]     + b_hh[Hn + c];
    float bixn = b_ih[2 * Hn + c];
    float bihn = b_hh[2 * Hn + c];
#pragma unroll
    for (int r = 0; r < 8; ++r) {
        int mm = mBase + r;
        size_t o = (size_t)mm * Hn + c;
        float rg = sigmoidf_(accR[r] + bihr);
        float zg = sigmoidf_(accZ[r] + bihz);
        float ng = tanhf(accNx[r] + bixn + rg * (accNh[r] + bihn));
        float hs = hsf[o];
        h[o] = (1.0f - zg) * ng + zg * hs;
    }
}

// ---------------- kernel 4: loss epilogue ----------------
__global__ void loss_kernel(const float* __restrict__ num, const float* __restrict__ den,
                            float* __restrict__ out) {
    __shared__ float sm[128];
    int tid = threadIdx.x;
    sm[tid] = num[tid] / (den[tid] + 1e-5f);
    __syncthreads();
    for (int s = 64; s > 0; s >>= 1) {
        if (tid < s) sm[tid] += sm[tid + s];
        __syncthreads();
    }
    if (tid == 0) out[(size_t)Bn * Tn * Vn] = sm[0];
}

// ---------------- launch ----------------
extern "C" void kernel_launch(void* const* d_in, const int* in_sizes, int n_in,
                              void* d_out, int out_size, void* d_ws, size_t ws_size,
                              hipStream_t stream) {
    const float* x       = (const float*)d_in[0];
    const float* mask    = (const float*)d_in[1];
    const float* deltas  = (const float*)d_in[2];
    const float* meanset = (const float*)d_in[3];
    const float* Wdh     = (const float*)d_in[4];
    const float* bdh     = (const float*)d_in[5];
    const float* Wdx     = (const float*)d_in[6];
    const float* bdx     = (const float*)d_in[7];
    const float* W_ih    = (const float*)d_in[8];
    const float* W_hh    = (const float*)d_in[9];
    const float* b_ih    = (const float*)d_in[10];
    const float* b_hh    = (const float*)d_in[11];
    float* out = (float*)d_out;

    char* ws = (char*)d_ws;
    unsigned short* amb   = (unsigned short*)(ws + AMB_OFF);
    unsigned short* db    = (unsigned short*)(ws + DB_OFF);
    unsigned short* wih_b = (unsigned short*)(ws + WIH_OFF);
    unsigned short* whh_b = (unsigned short*)(ws + WHH_OFF);
    unsigned short* wdh_b = (unsigned short*)(ws + WDH_OFF);
    float*          hbuf  = (float*)(ws + H_OFF);
    float*          hsf   = (float*)(ws + HSF_OFF);
    unsigned short* hs_b  = (unsigned short*)(ws + HSB_OFF);
    float*          num   = (float*)(ws + NUM_OFF);
    float*          den   = (float*)(ws + DEN_OFF);

    // zero padding regions, h state, loss accumulators (stream-ordered; capture-safe)
    size_t zn = (ws_size < WS_NEED) ? ws_size : WS_NEED;
    hipMemsetAsync(d_ws, 0, zn, stream);

    // weight conversion: 768*256 elems max
    weights_kernel<<<(G3H * Hn + 255) / 256, 256, 0, stream>>>(W_ih, W_hh, Wdh,
                                                               wih_b, whh_b, wdh_b);
    // elementwise scan over (b,v): 2048*59 = 120832 = 472 * 256 threads exactly
    scan_kernel<<<(Bn * Vn) / 256, 256, 0, stream>>>(x, mask, deltas, meanset, Wdx, bdx,
                                                     out, amb, db, num, den);
    // sequential recurrence: 2048 wave-tiles per kernel -> 256 blocks of 256 (8 waves)
    for (int t = 0; t < Tn; ++t) {
        stepA_kernel<<<256, 256, 0, stream>>>(db, wdh_b, bdh, hbuf, hsf, hs_b, t);
        stepB_kernel<<<256, 256, 0, stream>>>(amb, hs_b, wih_b, whh_b, b_ih, b_hh,
                                              hsf, hbuf, t);
    }
    loss_kernel<<<1, 128, 0, stream>>>(num, den, out);
}
